// WindowAttention_62345745269483
// MI455X (gfx1250) — matrix operand order repaired and checked by reading, not compiled, not run on hardware
//
#include <hip/hip_runtime.h>

#define HEADS   8
#define KD      64
#define DM      512
#define WIN     100
#define NWINS   41
#define NBATCH  8
#define NWG     (NBATCH*NWINS)   // 328
#define MPAD    112              // 7 tiles of 16
#define SPAD    128              // PV K-dim padded to 4x32
#define XSTR    520              // LDS stride (bf16 elems) for x/o staging
#define PSTR    136              // LDS stride for probs

typedef __attribute__((ext_vector_type(16))) __bf16 v16bf;
typedef __attribute__((ext_vector_type(8)))  float  v8f;

union ABu { uint4 q[2]; unsigned int u[8]; v16bf v; };

__device__ __forceinline__ unsigned short f2bf(float f) {
  unsigned int u = __float_as_uint(f);
  u += 0x7FFFu + ((u >> 16) & 1u);            // round-to-nearest-even
  return (unsigned short)(u >> 16);
}
__device__ __forceinline__ v8f zero8() {
  v8f z;
  #pragma unroll
  for (int i = 0; i < 8; ++i) z[i] = 0.0f;
  return z;
}

// ---------------- workspace layout (u16 elements) ----------------
// Wpk: 4 mats, [256 k-pairs][512 n][2]           = 1,048,576
// Q  : [328][112][512] row-major bf16            = 18,808,832
// Kpk: [328][8][32 kd-pairs][112 s][2]           = 18,808,832
// Vpk: [328][8][64 s-pairs][64 kd][2]            = 21,495,808
// O  : [328][112][512] row-major bf16            = 18,808,832
static constexpr size_t WPK_OFF = 0;
static constexpr size_t Q_OFF   = 1048576;
static constexpr size_t K_OFF   = Q_OFF + 18808832;
static constexpr size_t V_OFF   = K_OFF + 18808832;
static constexpr size_t O_OFF   = V_OFF + (size_t)NWG*HEADS*64*KD*2;

// ============== kernel 0: pack weights to bf16 B-operand layout ==============
__global__ void k_pack(const float* __restrict__ Wq, const float* __restrict__ Wk,
                       const float* __restrict__ Wv, const float* __restrict__ Wo,
                       unsigned short* __restrict__ wpk) {
  size_t i = (size_t)blockIdx.x * 256 + threadIdx.x;     // 4*512*512 total
  if (i >= (size_t)4 * DM * DM) return;
  int mat = (int)(i >> 18);
  int rem = (int)(i & 262143);
  int d = rem >> 9, c = rem & 511;                        // d = contraction row
  const float* W = (mat == 0) ? Wq : (mat == 1) ? Wk : (mat == 2) ? Wv : Wo;
  wpk[(size_t)mat*262144 + ((((size_t)(d >> 1))*DM + c) << 1) + (d & 1)] = f2bf(W[rem]);
}

// ============== kernel 1: QKV projections (1 block per window) ==============
__global__ void k_qkv(const float* __restrict__ x,
                      const unsigned int* __restrict__ wpk32,
                      const float* __restrict__ bq, const float* __restrict__ bk,
                      const float* __restrict__ bv,
                      unsigned short* __restrict__ qws,
                      unsigned short* __restrict__ kws,
                      unsigned short* __restrict__ vws) {
  extern __shared__ char smem[];
  unsigned short* xs = (unsigned short*)smem;             // [112][XSTR]
  const int w = blockIdx.x, b = w / NWINS, win = w % NWINS;
  const int tid = threadIdx.x;

  for (int i = tid; i < MPAD * DM; i += 256) {            // stage window, bf16
    int r = i >> 9, c = i & 511;
    int sr = win * WIN + r;
    float v = 0.0f;
    if (r < WIN && sr < 4096) v = x[((size_t)b*4096 + sr)*DM + c];
    xs[r*XSTR + c] = f2bf(v);
  }
  for (int i = tid; i < HEADS*8*KD*2; i += 256) {         // zero V pad s-pairs 56..63
    int h = i >> 10, off = i & 1023;
    vws[(((size_t)(w*HEADS + h)*64 + 56)*KD)*2 + off] = 0;
  }
  __syncthreads();

  const int wave = tid >> 5, lane = tid & 31;
  const int lh = lane >> 4, ln = lane & 15;

  for (int sidx = wave; sidx < 96; sidx += 8) {           // (mat, n-tile) strips
    const int mat = sidx >> 5, nt = sidx & 31;
    v8f acc[7];
    #pragma unroll
    for (int m = 0; m < 7; ++m) acc[m] = zero8();
    for (int kt = 0; kt < 16; ++kt) {
      ABu bvec;
      const unsigned int* bp = wpk32 + ((size_t)(mat*256 + kt*16 + lh*8))*DM + nt*16 + ln;
      #pragma unroll
      for (int r = 0; r < 8; ++r) bvec.u[r] = bp[(size_t)r * DM];
      #pragma unroll
      for (int m = 0; m < 7; ++m) {
        ABu a;
        const unsigned short* ap = xs + (m*16 + ln)*XSTR + kt*32 + lh*16;
        a.q[0] = *(const uint4*)ap;
        a.q[1] = *(const uint4*)(ap + 8);
        acc[m] = __builtin_amdgcn_wmma_f32_16x16x32_bf16(
            false, a.v, false, bvec.v, (short)0, acc[m], false, false);
      }
    }
    const int c = nt*16 + ln;
    const float bias = (mat == 0) ? bq[c] : (mat == 1) ? bk[c] : bv[c];
    const int h = c >> 6, kd = c & 63;
    #pragma unroll
    for (int m = 0; m < 7; ++m)
      #pragma unroll
      for (int v = 0; v < 8; ++v) {
        int row = m*16 + lh*8 + v;
        unsigned short val = f2bf(acc[m][v] + bias);
        if (mat == 0)      qws[((size_t)(w*MPAD + row))*DM + c] = val;
        else if (mat == 1) kws[(((size_t)(w*HEADS + h)*32 + (kd >> 1))*MPAD + row)*2 + (kd & 1)] = val;
        else               vws[(((size_t)(w*HEADS + h)*64 + (row >> 1))*KD + kd)*2 + (row & 1)] = val;
      }
  }
}

// ============== kernel 2: attention (1 block per window*head) ==============
__global__ void k_attn(const unsigned short* __restrict__ qws,
                       const unsigned int* __restrict__ kws32,
                       const unsigned int* __restrict__ vws32,
                       unsigned short* __restrict__ ows) {
  extern __shared__ char smem[];
  float* sf = (float*)smem;                                     // [112][128] f32
  unsigned short* pb = (unsigned short*)(smem + MPAD*SPAD*4);   // [112][PSTR] bf16
  const int w = blockIdx.x >> 3, h = blockIdx.x & 7;
  const int tid = threadIdx.x, wave = tid >> 5, lane = tid & 31;
  const int lh = lane >> 4, ln = lane & 15;

  for (int t = wave; t < 49; t += 8) {                    // S = Q K^T * scale
    int mt = t / 7, nt = t % 7;
    v8f acc = zero8();
    for (int ks = 0; ks < 2; ++ks) {
      ABu a, bu;
      const unsigned short* ap = qws + ((size_t)(w*MPAD + mt*16 + ln))*DM + h*KD + ks*32 + lh*16;
      a.q[0] = *(const uint4*)ap;
      a.q[1] = *(const uint4*)(ap + 8);
      const unsigned int* bp = kws32 + ((size_t)(w*HEADS + h)*32 + ks*16 + lh*8)*MPAD + nt*16 + ln;
      #pragma unroll
      for (int r = 0; r < 8; ++r) bu.u[r] = bp[(size_t)r * MPAD];
      acc = __builtin_amdgcn_wmma_f32_16x16x32_bf16(
          false, a.v, false, bu.v, (short)0, acc, false, false);
    }
    int s = nt*16 + ln;
    #pragma unroll
    for (int v = 0; v < 8; ++v) sf[(mt*16 + lh*8 + v)*SPAD + s] = acc[v] * 0.125f;
  }
  __syncthreads();

  if (tid < MPAD) {                                       // masked softmax per row
    const float* rp = sf + tid*SPAD;
    float mx = -1e30f;
    for (int s = 0; s < WIN; ++s) mx = fmaxf(mx, rp[s]);
    float sum = 0.0f;
    for (int s = 0; s < WIN; ++s) sum += __expf(rp[s] - mx);
    float inv = 1.0f / sum;
    unsigned short* pr = pb + tid*PSTR;
    for (int s = 0; s < SPAD; ++s)
      pr[s] = f2bf((s < WIN) ? __expf(rp[s] - mx) * inv : 0.0f);
  }
  __syncthreads();

  for (int t = wave; t < 28; t += 8) {                    // O = P V
    int mt = t >> 2, nt = t & 3;
    v8f acc = zero8();
    for (int ks = 0; ks < 4; ++ks) {
      ABu a, bu;
      const unsigned short* ap = pb + (mt*16 + ln)*PSTR + ks*32 + lh*16;
      a.q[0] = *(const uint4*)ap;
      a.q[1] = *(const uint4*)(ap + 8);
      const unsigned int* bp = vws32 + ((size_t)(w*HEADS + h)*64 + ks*16 + lh*8)*KD + nt*16 + ln;
      #pragma unroll
      for (int r = 0; r < 8; ++r) bu.u[r] = bp[(size_t)r * KD];
      acc = __builtin_amdgcn_wmma_f32_16x16x32_bf16(
          false, a.v, false, bu.v, (short)0, acc, false, false);
    }
    int kd = nt*16 + ln;
    #pragma unroll
    for (int v = 0; v < 8; ++v)
      ows[((size_t)(w*MPAD + mt*16 + lh*8 + v))*DM + h*KD + kd] = f2bf(acc[v]);
  }
}

// ============== kernel 3: output projection (1 block per window) ==============
__global__ void k_out(const unsigned short* __restrict__ ows,
                      const unsigned int* __restrict__ wpk32,
                      const float* __restrict__ bo,
                      float* __restrict__ out) {
  extern __shared__ char smem[];
  unsigned short* os = (unsigned short*)smem;             // [112][XSTR]
  const int w = blockIdx.x, tid = threadIdx.x;
  for (int i = tid; i < MPAD*(DM/8); i += 256) {          // stage O via uint4
    int r = i >> 6, c8 = i & 63;
    *(uint4*)(os + r*XSTR + c8*8) = *(const uint4*)(ows + ((size_t)(w*MPAD + r))*DM + c8*8);
  }
  __syncthreads();

  const int wave = tid >> 5, lane = tid & 31;
  const int lh = lane >> 4, ln = lane & 15;
  for (int nt = wave; nt < 32; nt += 8) {
    v8f acc[7];
    #pragma unroll
    for (int m = 0; m < 7; ++m) acc[m] = zero8();
    for (int kt = 0; kt < 16; ++kt) {
      ABu bvec;
      const unsigned int* bp = wpk32 + ((size_t)(3*256 + kt*16 + lh*8))*DM + nt*16 + ln;
      #pragma unroll
      for (int r = 0; r < 8; ++r) bvec.u[r] = bp[(size_t)r * DM];
      #pragma unroll
      for (int m = 0; m < 7; ++m) {
        ABu a;
        const unsigned short* ap = os + (m*16 + ln)*XSTR + kt*32 + lh*16;
        a.q[0] = *(const uint4*)ap;
        a.q[1] = *(const uint4*)(ap + 8);
        acc[m] = __builtin_amdgcn_wmma_f32_16x16x32_bf16(
            false, a.v, false, bvec.v, (short)0, acc[m], false, false);
      }
    }
    const int c = nt*16 + ln;
    const float bias = bo[c];
    #pragma unroll
    for (int m = 0; m < 7; ++m)
      #pragma unroll
      for (int v = 0; v < 8; ++v) {
        int row = m*16 + lh*8 + v;
        if (row < WIN) out[((size_t)(w*WIN + row))*DM + c] = acc[m][v] + bias;
      }
  }
}

extern "C" void kernel_launch(void* const* d_in, const int* in_sizes, int n_in,
                              void* d_out, int out_size, void* d_ws, size_t ws_size,
                              hipStream_t stream) {
  (void)in_sizes; (void)n_in; (void)out_size; (void)ws_size;
  const float* x  = (const float*)d_in[0];
  const float* Wq = (const float*)d_in[1];
  const float* bq = (const float*)d_in[2];
  const float* Wk = (const float*)d_in[3];
  const float* bk = (const float*)d_in[4];
  const float* Wv = (const float*)d_in[5];
  const float* bv = (const float*)d_in[6];
  const float* Wo = (const float*)d_in[7];
  const float* bo = (const float*)d_in[8];
  float* out = (float*)d_out;

  unsigned short* ws16 = (unsigned short*)d_ws;
  unsigned short* wpk = ws16 + WPK_OFF;
  unsigned short* qws = ws16 + Q_OFF;
  unsigned short* kws = ws16 + K_OFF;
  unsigned short* vws = ws16 + V_OFF;
  unsigned short* ows = ws16 + O_OFF;

  const int SMEM_BIG  = MPAD * XSTR * 2;                 // 116,480 B
  const int SMEM_ATTN = MPAD * SPAD * 4 + MPAD * PSTR * 2;  // 87,808 B
  (void)hipFuncSetAttribute((const void*)k_qkv,  hipFuncAttributeMaxDynamicSharedMemorySize, SMEM_BIG);
  (void)hipFuncSetAttribute((const void*)k_attn, hipFuncAttributeMaxDynamicSharedMemorySize, SMEM_ATTN);
  (void)hipFuncSetAttribute((const void*)k_out,  hipFuncAttributeMaxDynamicSharedMemorySize, SMEM_BIG);

  k_pack<<<4096, 256, 0, stream>>>(Wq, Wk, Wv, Wo, wpk);
  k_qkv<<<NWG, 256, SMEM_BIG, stream>>>(x, (const unsigned int*)wpk, bq, bk, bv, qws, kws, vws);
  k_attn<<<NWG * HEADS, 256, SMEM_ATTN, stream>>>(qws, (const unsigned int*)kws,
                                                  (const unsigned int*)vws, ows);
  k_out<<<NWG, 256, SMEM_BIG, stream>>>(ows, (const unsigned int*)wpk, bo, out);
}
